// DCVQQuantizer_17892833755580
// MI455X (gfx1250) — matrix operand order repaired and checked by reading, not compiled, not run on hardware
//
#include <hip/hip_runtime.h>

// ---------------------------------------------------------------------------
// DCVQ quantizer for MI455X (gfx1250, wave32, WMMA).
//   z:         [B=16, D=512, H=32, W=32] f32
//   codebooks: [N=8,  M=1024, ds=64]     f32
// Outputs (concatenated in d_out, f32):
//   out [16,512,32,32] | loss_vq | loss_commit | indices.T [16384, 8]
//
// Strategy: cross-term GEMM on bf16 WMMA with hi/lo split (fp32-grade argmin).
// Codebook is pre-split to bf16 hi/lo in workspace so the GEMM
// inner loop does only wide vector loads + v_wmma + argmin selects.
// ---------------------------------------------------------------------------

typedef __attribute__((ext_vector_type(16))) __bf16 v16bf;
typedef __attribute__((ext_vector_type(8)))  float  v8f;

#define N_CB   8
#define M_CB   1024
#define DS     64
#define T_TOK  16384             // 16 * 32 * 32
#define OUT_ELEMS 8388608        // 16*512*32*32 (== N*T*ds)
#define LOSS_OFF  OUT_ELEMS
#define IDX_OFF   (OUT_ELEMS + 2)

// workspace layout (bytes)
#define WS_C2_OFF   0                       // 8192 f32   = 32 KB
#define WS_BH_OFF   (8192 * 4)              // 524288 u16 = 1 MB
#define WS_BL_OFF   (WS_BH_OFF + 524288*2)  // 524288 u16 = 1 MB
#define WS_IDX_OFF  (WS_BL_OFF + 524288*2)  // 131072 i32 = 512 KB

static __device__ __forceinline__ unsigned short bf16_rne(float f) {
  unsigned int u = __float_as_uint(f);
  u += 0x7FFFu + ((u >> 16) & 1u);           // round-to-nearest-even
  return (unsigned short)(u >> 16);
}
static __device__ __forceinline__ float bf16_to_f(unsigned short h) {
  return __uint_as_float(((unsigned int)h) << 16);
}

union BfFrag { v16bf v; unsigned short s[16]; };

// ---------------------------------------------------------------------------
// Kernel 0a: c2[n,m] = ||codebook[n,m]||^2, zero loss accumulators.
// ---------------------------------------------------------------------------
__global__ __launch_bounds__(256) void vq_prep_c2(const float* __restrict__ cb,
                                                  float* __restrict__ c2,
                                                  float* __restrict__ loss_out) {
  int g = blockIdx.x * 256 + threadIdx.x;       // 0 .. 8191
  if (g == 0) { loss_out[0] = 0.0f; loss_out[1] = 0.0f; }
  if (g < N_CB * M_CB) {
    const float* p = cb + (size_t)g * DS;
    float s = 0.0f;
#pragma unroll
    for (int d = 0; d < DS; ++d) { float v = p[d]; s += v * v; }
    c2[g] = s;
  }
}

// ---------------------------------------------------------------------------
// Kernel 0b: hi/lo bf16 split of codebook:
//   cbt_hi/lo[n][m][k] = split(cb[n][m][k])   (fragment-ready: contiguous k)
// ---------------------------------------------------------------------------
__global__ __launch_bounds__(256) void vq_prep_cbt(const float* __restrict__ cb,
                                                   unsigned short* __restrict__ cbt_hi,
                                                   unsigned short* __restrict__ cbt_lo) {
  int g = blockIdx.x * 256 + threadIdx.x;       // n*64K + m*64 + k, 524288 total
  int m = (g >> 6) & 1023;
  int k = g & 63;
  int n = g >> 16;
  float f = cb[((size_t)(n * M_CB + m)) * DS + k];
  unsigned short hi = bf16_rne(f);
  cbt_hi[g] = hi;
  cbt_lo[g] = bf16_rne(f - bf16_to_f(hi));
}

// ---------------------------------------------------------------------------
// Kernel 1: per (n, 32-token tile): cross GEMM via WMMA bf16 hi/lo split,
// running argmin of (c2[m] - 2*cross) over all 1024 codes.
// One wave32 per block, two 16-row accumulator tiles sharing B fragments.
// ---------------------------------------------------------------------------
__global__ __launch_bounds__(32) void vq_argmin(const float* __restrict__ z,
                                                const unsigned short* __restrict__ cbt_hi,
                                                const unsigned short* __restrict__ cbt_lo,
                                                const float* __restrict__ c2,
                                                int* __restrict__ idx_ws) {
  const int lane = threadIdx.x;      // 0..31
  const int n    = blockIdx.y;       // codebook group
  const int t0   = blockIdx.x << 5;  // 32-row token tile base
  const int half = lane >> 4;        // 0 / 1
  const int l15  = lane & 15;
  const int kbase = half << 3;

  // ---- A fragments: z_flat[n, rows, 0..63], hi/lo split, 2 tiles x 2 ksteps.
  // A 16x32 layout used here (contiguous k, same as B): lane(l<16: row=l | l>=16: row=l-16);
  // element i -> k = kstep*32 + half*16 + i.
  v16bf ahi[2][2], alo[2][2];
#pragma unroll
  for (int tile = 0; tile < 2; ++tile) {
    int row = t0 + (tile << 4) + l15;
    int b   = row >> 10;
    int hw  = row & 1023;
#pragma unroll
    for (int ks = 0; ks < 2; ++ks) {
      BfFrag fh, fl;
#pragma unroll
      for (int i = 0; i < 16; ++i) {
        int d = ks * 32 + (half << 4) + i;
        float f = z[(((size_t)(b * 512 + n * 64 + d)) << 10) + hw];
        unsigned short hi = bf16_rne(f);
        fh.s[i] = hi;
        fl.s[i] = bf16_rne(f - bf16_to_f(hi));
      }
      ahi[tile][ks] = fh.v;
      alo[tile][ks] = fl.v;
    }
  }

  // ---- B fragment pointers: lane holds column m0 + (lane&15),
  // element i = k = kstep*32 + half*16 + i (contiguous in the pre-split layout).
  const int k0 = l15 * DS + (half << 4);
  const unsigned short* pbh0 = cbt_hi + (size_t)n * DS * M_CB + k0;
  const unsigned short* pbh1 = cbt_hi + (size_t)n * DS * M_CB + 32 + k0;
  const unsigned short* pbl0 = cbt_lo + (size_t)n * DS * M_CB + k0;
  const unsigned short* pbl1 = cbt_lo + (size_t)n * DS * M_CB + 32 + k0;
  const float* pc2 = c2 + n * M_CB + l15;

  float best[2][8];
  int   bidx[2][8];
#pragma unroll
  for (int t = 0; t < 2; ++t)
#pragma unroll
    for (int r = 0; r < 8; ++r) { best[t][r] = 3.0e38f; bidx[t][r] = 0; }

  for (int m0 = 0; m0 < M_CB; m0 += 16) {
    v16bf bh0 = *(const v16bf*)(pbh0 + m0 * DS);
    v16bf bh1 = *(const v16bf*)(pbh1 + m0 * DS);
    v16bf bl0 = *(const v16bf*)(pbl0 + m0 * DS);
    v16bf bl1 = *(const v16bf*)(pbl1 + m0 * DS);
    float c2v = pc2[m0];
    int   mcol = m0 + l15;

#pragma unroll
    for (int t = 0; t < 2; ++t) {
      v8f acc = {0.f, 0.f, 0.f, 0.f, 0.f, 0.f, 0.f, 0.f};
      // fp32-grade cross: hi*hi + hi*lo + lo*hi, both K-steps
      acc = __builtin_amdgcn_wmma_f32_16x16x32_bf16(false, ahi[t][0], false, bh0, (short)0, acc, false, false);
      acc = __builtin_amdgcn_wmma_f32_16x16x32_bf16(false, ahi[t][0], false, bl0, (short)0, acc, false, false);
      acc = __builtin_amdgcn_wmma_f32_16x16x32_bf16(false, alo[t][0], false, bh0, (short)0, acc, false, false);
      acc = __builtin_amdgcn_wmma_f32_16x16x32_bf16(false, ahi[t][1], false, bh1, (short)0, acc, false, false);
      acc = __builtin_amdgcn_wmma_f32_16x16x32_bf16(false, ahi[t][1], false, bl1, (short)0, acc, false, false);
      acc = __builtin_amdgcn_wmma_f32_16x16x32_bf16(false, alo[t][1], false, bh1, (short)0, acc, false, false);

      // C/D layout: lane -> col = lane&15, VGPR r -> row = r + half*8.
#pragma unroll
      for (int r = 0; r < 8; ++r) {
        float dist = c2v - 2.0f * acc[r];       // z^2 term constant per row
        if (dist < best[t][r]) { best[t][r] = dist; bidx[t][r] = mcol; }
      }
    }
  }

  // ---- min-reduce across the 16 lanes holding one row's columns.
#pragma unroll
  for (int t = 0; t < 2; ++t) {
#pragma unroll
    for (int r = 0; r < 8; ++r) {
      float v  = best[t][r];
      int   ix = bidx[t][r];
#pragma unroll
      for (int s = 1; s < 16; s <<= 1) {
        float ov = __shfl_xor(v,  s, 32);
        int   oi = __shfl_xor(ix, s, 32);
        if (ov < v || (ov == v && oi < ix)) { v = ov; ix = oi; }
      }
      if (l15 == 0) {
        int trow = t0 + (t << 4) + (half << 3) + r;
        idx_ws[n * T_TOK + trow] = ix;
      }
    }
  }
}

// ---------------------------------------------------------------------------
// Kernel 2: gather codes -> out (coalesced over hw), MSE losses, index map.
// ---------------------------------------------------------------------------
__global__ __launch_bounds__(256) void vq_gather(const float* __restrict__ z,
                                                 const float* __restrict__ cb,
                                                 const int* __restrict__ idx_ws,
                                                 float* __restrict__ out) {
  __shared__ float red[256];
  int g  = blockIdx.x * 256 + threadIdx.x;  // 0 .. 131071 == N * T
  int n  = g >> 14;
  int t  = g & (T_TOK - 1);
  int b  = t >> 10;
  int hw = t & 1023;

  int idx = idx_ws[n * T_TOK + t];
  out[IDX_OFF + (size_t)t * N_CB + n] = (float)idx;     // indices.T as f32

  const float* code = cb + ((size_t)(n * M_CB + idx)) * DS;
  float acc = 0.0f;
#pragma unroll 4
  for (int d = 0; d < DS; ++d) {
    size_t a  = (((size_t)(b * 512 + n * 64 + d)) << 10) + hw;
    float  c  = code[d];
    float  zv = z[a];
    out[a] = c;                                         // straight-through fwd
    float df = c - zv;
    acc += df * df;
  }

  red[threadIdx.x] = acc;
  __syncthreads();
  for (int s = 128; s > 0; s >>= 1) {
    if (threadIdx.x < s) red[threadIdx.x] += red[threadIdx.x + s];
    __syncthreads();
  }
  if (threadIdx.x == 0) {
    float part = red[0] * (1.0f / (float)OUT_ELEMS);    // mean over N*T*ds
    atomicAdd(&out[LOSS_OFF + 0], part);                // loss_vq
    atomicAdd(&out[LOSS_OFF + 1], part);                // loss_commit (same fwd)
  }
}

// ---------------------------------------------------------------------------
extern "C" void kernel_launch(void* const* d_in, const int* in_sizes, int n_in,
                              void* d_out, int out_size, void* d_ws, size_t ws_size,
                              hipStream_t stream) {
  const float* z  = (const float*)d_in[0];   // [16,512,32,32]
  const float* cb = (const float*)d_in[1];   // [8,1024,64]
  float* out = (float*)d_out;

  float*          c2     = (float*)((char*)d_ws + WS_C2_OFF);
  unsigned short* cbt_hi = (unsigned short*)((char*)d_ws + WS_BH_OFF);
  unsigned short* cbt_lo = (unsigned short*)((char*)d_ws + WS_BL_OFF);
  int*            idx_ws = (int*)((char*)d_ws + WS_IDX_OFF);

  vq_prep_c2 <<<32,   256, 0, stream>>>(cb, c2, out + LOSS_OFF);
  vq_prep_cbt<<<2048, 256, 0, stream>>>(cb, cbt_hi, cbt_lo);
  vq_argmin  <<<dim3(T_TOK / 32, N_CB), 32, 0, stream>>>(z, cbt_hi, cbt_lo, c2, idx_ws);
  vq_gather  <<<(N_CB * T_TOK) / 256, 256, 0, stream>>>(z, cb, idx_ws, out);
}
